// QWKLoss_83348135346706
// MI455X (gfx1250) — 8-bit matrix operand order repaired and checked by reading, compiled, not run on hardware
//
#include <hip/hip_runtime.h>


typedef int v8i __attribute__((ext_vector_type(8)));

#define HIST_BLOCKS 512
#define HIST_TPB    256

__global__ void qwk_zero_kernel(unsigned* __restrict__ cm) {
    if (threadIdx.x < 256) cm[threadIdx.x] = 0u;
}

__device__ __forceinline__ unsigned onehot4(unsigned d, unsigned rowpat) {
    const unsigned y = d ^ rowpat;
    const unsigned z = y - 0x01010101u;
    const unsigned w = z & ~y;
    return (w >> 7) & 0x01010101u;
}

__device__ __forceinline__ void sample_bytes(const float* __restrict__ logits,
                                             const int*   __restrict__ targets,
                                             int idx, int total,
                                             unsigned& tb, unsigned& pb) {
    const bool valid = idx < total;
    const int  li    = valid ? idx : 0;

    const float4* lp = (const float4*)(logits + (size_t)li * 8);
    const float4 l0 = lp[0];
    const float4 l1 = lp[1];

    float m = l0.x; int p = 0;
    bool g;
    g = l0.y > m; p = g ? 1 : p; m = g ? l0.y : m;
    g = l0.z > m; p = g ? 2 : p; m = g ? l0.z : m;
    g = l0.w > m; p = g ? 3 : p; m = g ? l0.w : m;
    g = l1.x > m; p = g ? 4 : p; m = g ? l1.x : m;
    g = l1.y > m; p = g ? 5 : p; m = g ? l1.y : m;
    g = l1.z > m; p = g ? 6 : p; m = g ? l1.z : m;
    g = l1.w > m; p = g ? 7 : p; m = g ? l1.w : m;

    const int t = targets[li];
    tb = (valid && (t > 0)) ? (unsigned)t : 0xFFu;
    pb = valid ? (unsigned)p : 0xFFu;
}

__global__ __launch_bounds__(HIST_TPB)
void qwk_hist_kernel(const float* __restrict__ logits,
                     const int*   __restrict__ targets,
                     unsigned*    __restrict__ cm_g,
                     int total) {
    __shared__ unsigned long long stageq[HIST_TPB / 32][16];
    __shared__ int csum[256];

    const int tid  = threadIdx.x;
    const int lane = tid & 31;
    const int wib  = tid >> 5;

    for (int i = tid; i < 256; i += HIST_TPB) csum[i] = 0;
    __syncthreads();

    const int      hsel   = lane >> 4;
    const unsigned rowpat = (unsigned)(lane & 15) * 0x01010101u;

    unsigned char* tstage = (unsigned char*)&stageq[wib][0];
    unsigned char* pstage = (unsigned char*)&stageq[wib][8];

    v8i c = {};

    const int nchunks = (total + 63) >> 6;
    const int gw = (int)((blockIdx.x * HIST_TPB + tid) >> 5);
    const int nw = (int)((gridDim.x  * HIST_TPB) >> 5);

    for (int ch = gw; ch < nchunks; ch += nw) {
        const int base = ch << 6;

        unsigned tb0, pb0, tb1, pb1;
        sample_bytes(logits, targets, base + lane,      total, tb0, pb0);
        sample_bytes(logits, targets, base + lane + 32, total, tb1, pb1);

        tstage[lane]      = (unsigned char)tb0;
        tstage[lane + 32] = (unsigned char)tb1;
        pstage[lane]      = (unsigned char)pb0;
        pstage[lane + 32] = (unsigned char)pb1;

        asm volatile("s_wait_dscnt 0x0" ::: "memory");

        const unsigned long long qA0 = stageq[wib][0 + hsel];
        const unsigned long long qA1 = stageq[wib][2 + hsel];
        const unsigned long long qA2 = stageq[wib][4 + hsel];
        const unsigned long long qA3 = stageq[wib][6 + hsel];
        const unsigned long long qB0 = stageq[wib][8  + hsel];
        const unsigned long long qB1 = stageq[wib][10 + hsel];
        const unsigned long long qB2 = stageq[wib][12 + hsel];
        const unsigned long long qB3 = stageq[wib][14 + hsel];

        asm volatile("" ::: "memory");

        v8i a, b;
        a[0] = (int)onehot4((unsigned)qA0,         rowpat);
        a[1] = (int)onehot4((unsigned)(qA0 >> 32), rowpat);
        a[2] = (int)onehot4((unsigned)qA1,         rowpat);
        a[3] = (int)onehot4((unsigned)(qA1 >> 32), rowpat);
        a[4] = (int)onehot4((unsigned)qA2,         rowpat);
        a[5] = (int)onehot4((unsigned)(qA2 >> 32), rowpat);
        a[6] = (int)onehot4((unsigned)qA3,         rowpat);
        a[7] = (int)onehot4((unsigned)(qA3 >> 32), rowpat);

        b[0] = (int)onehot4((unsigned)qB0,         rowpat);
        b[1] = (int)onehot4((unsigned)(qB0 >> 32), rowpat);
        b[2] = (int)onehot4((unsigned)qB1,         rowpat);
        b[3] = (int)onehot4((unsigned)(qB1 >> 32), rowpat);
        b[4] = (int)onehot4((unsigned)qB2,         rowpat);
        b[5] = (int)onehot4((unsigned)(qB2 >> 32), rowpat);
        b[6] = (int)onehot4((unsigned)qB3,         rowpat);
        b[7] = (int)onehot4((unsigned)(qB3 >> 32), rowpat);

        c = __builtin_amdgcn_wmma_i32_16x16x64_iu8(
             false, a,  false, b, c,
             false,  false);
    }

#pragma unroll
    for (int r = 0; r < 8; ++r) atomicAdd(&csum[r * 32 + lane], c[r]);
    __syncthreads();

    for (int i = tid; i < 256; i += HIST_TPB) {
        const int r  = i >> 5;
        const int ln = i & 31;
        const int M  = r + ((ln >> 4) << 3);
        const int N  = ln & 15;
        const unsigned v = (unsigned)csum[i];
        if (v) atomicAdd(&cm_g[M * 16 + N], v);
    }
}

__global__ void qwk_final_kernel(const unsigned* __restrict__ cm_g,
                                 float* __restrict__ out) {
    if (threadIdx.x != 0 || blockIdx.x != 0) return;
    float cm[8][8];
    float n = 0.0f;
    for (int i = 0; i < 8; ++i)
        for (int j = 0; j < 8; ++j) {
            const float v = (float)cm_g[i * 16 + j];
            cm[i][j] = v;
            n += v;
        }
    float loss = 0.0f;
    if (n > 0.0f) {
        const float inv = 1.0f / n;
        float mt[8], mp[8];
        for (int i = 0; i < 8; ++i) { mt[i] = 0.0f; mp[i] = 0.0f; }
        for (int i = 0; i < 8; ++i)
            for (int j = 0; j < 8; ++j) {
                const float pij = cm[i][j] * inv;
                cm[i][j] = pij;
                mt[i] += pij;
                mp[j] += pij;
            }
        float num = 0.0f, den = 0.0f;
        for (int i = 0; i < 8; ++i)
            for (int j = 0; j < 8; ++j) {
                const float d = (float)(i - j);
                const float w = 1.0f - d * d * (1.0f / 49.0f);
                num += w * cm[i][j];
                den += w * mt[i] * mp[j];
            }
        const float qwk = (den == 0.0f) ? 0.0f : num / den;
        loss = 1.0f - qwk;
    }
    out[0] = loss;
}

extern "C" void kernel_launch(void* const* d_in, const int* in_sizes, int n_in,
                              void* d_out, int out_size, void* d_ws, size_t ws_size,
                              hipStream_t stream) {
    const float* logits  = (const float*)d_in[0];
    const int*   targets = (const int*)d_in[1];
    float*       out     = (float*)d_out;
    unsigned*    cm      = (unsigned*)d_ws;

    const int total = in_sizes[1];

    qwk_zero_kernel<<<1, 256, 0, stream>>>(cm);
    qwk_hist_kernel<<<HIST_BLOCKS, HIST_TPB, 0, stream>>>(logits, targets, cm, total);
    qwk_final_kernel<<<1, 32, 0, stream>>>(cm, out);
}
